// VectorQuantizer_45775761441265
// MI455X (gfx1250) — compile-verified, matrix operand order checked by reading, not run on hardware
//
#include <hip/hip_runtime.h>
#include <hip/hip_bf16.h>
#include <hip/hip_fp16.h>

typedef __attribute__((ext_vector_type(16))) _Float16 v16h;
typedef __attribute__((ext_vector_type(8)))  float    v8f;
typedef __attribute__((ext_vector_type(2)))  float    v2f;

#define N_TOT   131072
#define DDIM    64
#define KCODES  1024
#define ROWS    128            // rows per block
#define CHUNK   64             // codes per streamed chunk
#define NCHUNK  (KCODES / CHUNK)
#define NWAVES  8
#define XPAD    72             // padded f16 row to dodge bank conflicts

// workspace layout (floats)
#define WS_COUNTS 0
#define WS_DW     (KCODES)                     // K*D floats
#define WS_SSE    (KCODES + KCODES * DDIM)     // 1 float
#define WS_EN     (WS_SSE + 1)                 // K floats of ||e||^2
#define WS_TOTAL  (WS_EN + KCODES)

// ---------------------------------------------------------------------------
// Kernel 0: zero accumulators, precompute per-code squared norms
// ---------------------------------------------------------------------------
__global__ void vq_init_kernel(const float* __restrict__ emb,
                               float* __restrict__ ws) {
    int i = blockIdx.x * blockDim.x + threadIdx.x;
    if (i < WS_EN) ws[i] = 0.0f;
    if (i < KCODES) {
        const float* e = emb + (size_t)i * DDIM;
        float s = 0.0f;
#pragma unroll 8
        for (int d = 0; d < DDIM; ++d) s += e[d] * e[d];
        ws[WS_EN + i] = s;
    }
}

// ---------------------------------------------------------------------------
// Kernel 1: fused scores -> softmax -> quant / counts / dw
//   Pass A: per-lane score max   (WMMA, no shuffles in loop)
//   Pass B: per-lane exp-sum     (WMMA, 1 exp/elem, no shuffles in loop)
//   Pass C: enc -> quant / counts / dw (WMMA f16 + WMMA f32)
// ---------------------------------------------------------------------------
__launch_bounds__(256)
__global__ void vq_main_kernel(const float* __restrict__ x,
                               const float* __restrict__ emb,
                               const float* __restrict__ temp,
                               float* __restrict__ out,
                               float* __restrict__ ws) {
    __shared__ _Float16 sX[ROWS][XPAD];          // x tile (f16)
    __shared__ _Float16 sE[CHUNK][XPAD];         // embedding chunk (f16)
    __shared__ float    sEn[CHUNK];              // ||e||^2 chunk
    __shared__ float    sEnc[NWAVES][16][CHUNK]; // per-wave enc strip

    const int tid      = threadIdx.x;
    const int lane     = tid & 31;
    const int wave     = tid >> 5;
    const int laneIn   = lane & 15;
    const int laneHalf = lane >> 4;
    const int row0     = wave * 16;                  // strip base in block
    const int blockRow = blockIdx.x * ROWS;

    const float invT  = 1.0f / temp[0];
    const float twoT  = 2.0f * invT;
    float*       cnt = ws + WS_COUNTS;
    float*       dw  = ws + WS_DW;
    const float* en  = ws + WS_EN;

    // ---- x tile -> LDS (f16) ----
    for (int i = tid; i < ROWS * DDIM; i += 256) {
        int r = i >> 6, d = i & 63;
        sX[r][d] = (_Float16)x[(size_t)(blockRow + r) * DDIM + d];
    }
    __syncthreads();

    // ---- A operand: this wave's 16-row strip in WMMA A layout (two K-steps)
    v16h a0, a1;
#pragma unroll
    for (int j = 0; j < 16; ++j) {
        int kk = ((j >> 3) << 4) + (laneHalf << 3) + (j & 7);
        a0[j] = sX[row0 + laneIn][kk];
        a1[j] = sX[row0 + laneIn][32 + kk];
    }

    float m[8];
#pragma unroll
    for (int r = 0; r < 8; ++r) m[r] = -1e30f;

    // ======================= PASS A: per-lane max ==========================
    for (int c = 0; c < NCHUNK; ++c) {
        __syncthreads();
        for (int i = tid; i < CHUNK * DDIM; i += 256) {
            int r = i >> 6, d = i & 63;
            sE[r][d] = (_Float16)emb[(size_t)(c * CHUNK + r) * DDIM + d];
        }
        if (tid < CHUNK) sEn[tid] = en[c * CHUNK + tid];
        __syncthreads();
        if (c + 1 < NCHUNK)   // global_prefetch_b8 of next chunk
            __builtin_prefetch(emb + (size_t)((c + 1) * CHUNK) * DDIM, 0, 1);

#pragma unroll
        for (int t = 0; t < CHUNK / 16; ++t) {
            v16h b0, b1;     // B operand: E^T tile (codes along N)
#pragma unroll
            for (int j = 0; j < 16; ++j) {
                int kk = ((j >> 3) << 4) + (laneHalf << 3) + (j & 7);
                b0[j] = sE[t * 16 + laneIn][kk];
                b1[j] = sE[t * 16 + laneIn][32 + kk];
            }
            v8f acc = {};
            acc = __builtin_amdgcn_wmma_f32_16x16x32_f16(false, a0, false, b0,
                                                         (short)0, acc, false, false);
            acc = __builtin_amdgcn_wmma_f32_16x16x32_f16(false, a1, false, b1,
                                                         (short)0, acc, false, false);
            float eNT = sEn[t * 16 + laneIn] * invT;
#pragma unroll
            for (int r = 0; r < 8; ++r)
                m[r] = fmaxf(m[r], fmaf(acc[r], twoT, -eNT));
        }
    }
    // merge per-lane maxes across the 16 lanes of this half
#pragma unroll
    for (int r = 0; r < 8; ++r) {
        m[r] = fmaxf(m[r], __shfl_xor(m[r], 1));
        m[r] = fmaxf(m[r], __shfl_xor(m[r], 2));
        m[r] = fmaxf(m[r], __shfl_xor(m[r], 4));
        m[r] = fmaxf(m[r], __shfl_xor(m[r], 8));
    }

    float l[8];
#pragma unroll
    for (int r = 0; r < 8; ++r) l[r] = 0.0f;

    // ======================= PASS B: per-lane exp-sum ======================
    for (int c = 0; c < NCHUNK; ++c) {
        __syncthreads();
        for (int i = tid; i < CHUNK * DDIM; i += 256) {
            int r = i >> 6, d = i & 63;
            sE[r][d] = (_Float16)emb[(size_t)(c * CHUNK + r) * DDIM + d];
        }
        if (tid < CHUNK) sEn[tid] = en[c * CHUNK + tid];
        __syncthreads();

#pragma unroll
        for (int t = 0; t < CHUNK / 16; ++t) {
            v16h b0, b1;
#pragma unroll
            for (int j = 0; j < 16; ++j) {
                int kk = ((j >> 3) << 4) + (laneHalf << 3) + (j & 7);
                b0[j] = sE[t * 16 + laneIn][kk];
                b1[j] = sE[t * 16 + laneIn][32 + kk];
            }
            v8f acc = {};
            acc = __builtin_amdgcn_wmma_f32_16x16x32_f16(false, a0, false, b0,
                                                         (short)0, acc, false, false);
            acc = __builtin_amdgcn_wmma_f32_16x16x32_f16(false, a1, false, b1,
                                                         (short)0, acc, false, false);
            float eNT = sEn[t * 16 + laneIn] * invT;
#pragma unroll
            for (int r = 0; r < 8; ++r)
                l[r] += __expf(fmaf(acc[r], twoT, -eNT) - m[r]);
        }
    }
    // merge per-lane sums, invert
    float rl[8];
#pragma unroll
    for (int r = 0; r < 8; ++r) {
        l[r] += __shfl_xor(l[r], 1);
        l[r] += __shfl_xor(l[r], 2);
        l[r] += __shfl_xor(l[r], 4);
        l[r] += __shfl_xor(l[r], 8);
        rl[r] = 1.0f / l[r];
    }

    v8f qacc[4] = {};   // quant strip: 4 d-tiles x (16 rows x 16 cols)

    // =============== PASS C: enc -> quant / counts / dw ====================
    for (int c = 0; c < NCHUNK; ++c) {
        __syncthreads();
        for (int i = tid; i < CHUNK * DDIM; i += 256) {
            int r = i >> 6, d = i & 63;
            sE[r][d] = (_Float16)emb[(size_t)(c * CHUNK + r) * DDIM + d];
        }
        if (tid < CHUNK) sEn[tid] = en[c * CHUNK + tid];
        __syncthreads();

#pragma unroll
        for (int t = 0; t < CHUNK / 16; ++t) {
            v16h b0, b1;
#pragma unroll
            for (int j = 0; j < 16; ++j) {
                int kk = ((j >> 3) << 4) + (laneHalf << 3) + (j & 7);
                b0[j] = sE[t * 16 + laneIn][kk];
                b1[j] = sE[t * 16 + laneIn][32 + kk];
            }
            v8f acc = {};
            acc = __builtin_amdgcn_wmma_f32_16x16x32_f16(false, a0, false, b0,
                                                         (short)0, acc, false, false);
            acc = __builtin_amdgcn_wmma_f32_16x16x32_f16(false, a1, false, b1,
                                                         (short)0, acc, false, false);
            float eNT = sEn[t * 16 + laneIn] * invT;
            float csum = 0.0f;
#pragma unroll
            for (int r = 0; r < 8; ++r) {
                float encv = __expf(fmaf(acc[r], twoT, -eNT) - m[r]) * rl[r];
                csum += encv;
                sEnc[wave][(laneHalf << 3) + r][t * 16 + laneIn] = encv;
            }
            csum += __shfl_xor(csum, 16);            // both halves (16 rows)
            if (laneHalf == 0)
                atomicAdd(&cnt[c * CHUNK + t * 16 + laneIn], csum);
        }

        // ---- quant += enc_strip (16 x CHUNK) @ E_chunk (CHUNK x 64) ----
#pragma unroll
        for (int ks = 0; ks < CHUNK / 32; ++ks) {
            v16h ae;
#pragma unroll
            for (int j = 0; j < 16; ++j) {
                int kk = ks * 32 + ((j >> 3) << 4) + (laneHalf << 3) + (j & 7);
                ae[j] = (_Float16)sEnc[wave][laneIn][kk];
            }
#pragma unroll
            for (int dt = 0; dt < 4; ++dt) {
                v16h be;
#pragma unroll
                for (int j = 0; j < 16; ++j) {
                    int kk = ks * 32 + ((j >> 3) << 4) + (laneHalf << 3) + (j & 7);
                    be[j] = sE[kk][dt * 16 + laneIn];
                }
                qacc[dt] = __builtin_amdgcn_wmma_f32_16x16x32_f16(
                    false, ae, false, be, (short)0, qacc[dt], false, false);
            }
        }

        // ---- dw += enc^T (codes x 16 rows) @ x_strip (16 rows x 64) ----
#pragma unroll
        for (int ct = 0; ct < CHUNK / 16; ++ct) {
#pragma unroll
            for (int dt = 0; dt < 4; ++dt) {
                v8f dacc = {};
#if __has_builtin(__builtin_amdgcn_wmma_f32_16x16x4_f32)
#pragma unroll
                for (int rs = 0; rs < 4; ++rs) {     // 16 rows, K=4 per step
                    int kr = rs * 4 + laneHalf * 2;
                    v2f av, bv;
                    av.x = sEnc[wave][kr + 0][ct * 16 + laneIn];
                    av.y = sEnc[wave][kr + 1][ct * 16 + laneIn];
                    bv.x = (float)sX[row0 + kr + 0][dt * 16 + laneIn];
                    bv.y = (float)sX[row0 + kr + 1][dt * 16 + laneIn];
                    dacc = __builtin_amdgcn_wmma_f32_16x16x4_f32(
                        false, av, false, bv, (short)0, dacc, false, false);
                }
#else
#pragma unroll
                for (int r = 0; r < 8; ++r) {
                    float a2 = 0.0f;
#pragma unroll
                    for (int rr = 0; rr < 16; ++rr)
                        a2 += sEnc[wave][rr][ct * 16 + (laneHalf << 3) + r] *
                              (float)sX[row0 + rr][dt * 16 + laneIn];
                    dacc[r] = a2;
                }
#endif
#pragma unroll
                for (int r = 0; r < 8; ++r) {
                    int code = c * CHUNK + ct * 16 + (laneHalf << 3) + r;
                    atomicAdd(&dw[(size_t)code * DDIM + dt * 16 + laneIn], dacc[r]);
                }
            }
        }
    }

    // ---- write quant (straight-through == quant) + accumulate SSE ----
    float ssep = 0.0f;
#pragma unroll
    for (int dt = 0; dt < 4; ++dt) {
#pragma unroll
        for (int r = 0; r < 8; ++r) {
            int grow = blockRow + row0 + (laneHalf << 3) + r;
            int gcol = dt * 16 + laneIn;
            float q  = qacc[dt][r];
            float xv = x[(size_t)grow * DDIM + gcol];
            float dq = q - xv;
            ssep += dq * dq;
            out[(size_t)grow * DDIM + gcol] = xv + dq;  // == q
        }
    }
    ssep += __shfl_xor(ssep, 1);
    ssep += __shfl_xor(ssep, 2);
    ssep += __shfl_xor(ssep, 4);
    ssep += __shfl_xor(ssep, 8);
    ssep += __shfl_xor(ssep, 16);
    if (lane == 0) atomicAdd(&ws[WS_SSE], ssep);
}

// ---------------------------------------------------------------------------
// Kernel 2: EMA updates, reg / entropy / diversity, loss & perplexity
// ---------------------------------------------------------------------------
__launch_bounds__(1024)
__global__ void vq_final_kernel(const float* __restrict__ ema_cs,
                                const float* __restrict__ ema_w,
                                const float* __restrict__ usage,
                                const float* __restrict__ ws,
                                float* __restrict__ out) {
    __shared__ float red[1024];
    const int k = threadIdx.x;

    float counts = ws[WS_COUNTS + k];
    float new_cs = 0.99f * ema_cs[k] + 0.01f * counts;

    red[k] = new_cs;
    __syncthreads();
    for (int s = 512; s > 0; s >>= 1) {
        if (k < s) red[k] += red[k + s];
        __syncthreads();
    }
    float nsum = red[0];
    __syncthreads();

    float cs     = (new_cs + 1e-5f) / (nsum + 1024.0f * 1e-5f) * nsum;
    float inv_cs = 1.0f / cs;
    float regp   = 0.0f;
#pragma unroll 8
    for (int d = 0; d < DDIM; ++d) {
        float w = 0.99f * ema_w[(size_t)k * DDIM + d] +
                  0.01f * ws[WS_DW + (size_t)k * DDIM + d];
        float e = w * inv_cs;
        regp += e * e;
    }

    float avg  = counts / (float)N_TOT;
    float entp = -avg * __logf(avg + 1e-10f);
    float nu   = 0.99f * usage[k] + 0.01f * counts;

    red[k] = nu;
    __syncthreads();
    for (int s = 512; s > 0; s >>= 1) {
        if (k < s) red[k] += red[k + s];
        __syncthreads();
    }
    float usum = red[0];
    __syncthreads();

    float p    = nu / (usum + 1e-5f);
    float divp = -p * __logf(p + 1e-10f);

    red[k] = regp;
    __syncthreads();
    for (int s = 512; s > 0; s >>= 1) {
        if (k < s) red[k] += red[k + s];
        __syncthreads();
    }
    float reg_total = red[0];
    __syncthreads();

    red[k] = entp;
    __syncthreads();
    for (int s = 512; s > 0; s >>= 1) {
        if (k < s) red[k] += red[k + s];
        __syncthreads();
    }
    float ent_total = red[0];
    __syncthreads();

    red[k] = divp;
    __syncthreads();
    for (int s = 512; s > 0; s >>= 1) {
        if (k < s) red[k] += red[k + s];
        __syncthreads();
    }
    float div_total = red[0];

    if (k == 0) {
        float mse  = ws[WS_SSE] / (float)((size_t)N_TOT * DDIM);
        float loss = mse + 0.25f * mse + reg_total +
                     0.8f * (ent_total + div_total);
        out[(size_t)N_TOT * DDIM + 0] = loss;
        out[(size_t)N_TOT * DDIM + 1] = __expf(ent_total);
    }
}

// ---------------------------------------------------------------------------
extern "C" void kernel_launch(void* const* d_in, const int* in_sizes, int n_in,
                              void* d_out, int out_size, void* d_ws, size_t ws_size,
                              hipStream_t stream) {
    const float* x      = (const float*)d_in[0];
    const float* emb    = (const float*)d_in[1];
    const float* temp   = (const float*)d_in[2];
    const float* ema_cs = (const float*)d_in[3];
    const float* ema_w  = (const float*)d_in[4];
    const float* usage  = (const float*)d_in[5];
    float* out = (float*)d_out;
    float* ws  = (float*)d_ws;

    vq_init_kernel<<<(WS_EN + 255) / 256, 256, 0, stream>>>(emb, ws);
    vq_main_kernel<<<N_TOT / ROWS, 256, 0, stream>>>(x, emb, temp, out, ws);
    vq_final_kernel<<<1, 1024, 0, stream>>>(ema_cs, ema_w, usage, ws, out);
}
